// MoEMLP_3848290697277
// MI455X (gfx1250) — matrix operand order repaired and checked by reading, not compiled, not run on hardware
//
#include <hip/hip_runtime.h>
#include <stdint.h>

#define DIM 1024
#define EXPERTS 8
#define HID 512
#define MT 32                 // tokens per block tile
#define LSTR 520              // padded LDS row stride in halfs (1040 B = 16*65 -> conflict-free)

typedef __attribute__((ext_vector_type(16))) __bf16 v16bf;
typedef __attribute__((ext_vector_type(8)))  float  v8f;

union BFrag { uint4 q[2]; v16bf bf; };

__device__ __forceinline__ unsigned short f2bf(float f) {
    unsigned int u = __builtin_bit_cast(unsigned int, f);
    unsigned int r = u + 0x7FFFu + ((u >> 16) & 1u);   // round-to-nearest-even
    return (unsigned short)(r >> 16);
}

// ---------------- kernel 1: zero counts + cursors (16 ints) ----------------
__global__ void moe_init_kernel(int* counts) {
    if (threadIdx.x < 16) counts[threadIdx.x] = 0;
}

// ---------------- kernel 2: fp32 -> bf16 weight conversion -----------------
__global__ void moe_cvt_kernel(const float* __restrict__ src,
                               unsigned short* __restrict__ dst, int n4) {
    int i = blockIdx.x * blockDim.x + threadIdx.x;
    if (i >= n4) return;
    float4 f = ((const float4*)src)[i];
    unsigned int lo = (unsigned int)f2bf(f.x) | ((unsigned int)f2bf(f.y) << 16);
    unsigned int hi = (unsigned int)f2bf(f.z) | ((unsigned int)f2bf(f.w) << 16);
    ((uint2*)dst)[i] = make_uint2(lo, hi);
}

// ---------------- kernel 3: router (one wave32 per token) ------------------
__global__ void moe_router_kernel(const float* __restrict__ x,
                                  const float* __restrict__ wr,
                                  int* __restrict__ eid,
                                  int* __restrict__ counts, int ntok) {
    int wave = (int)((blockIdx.x * blockDim.x + threadIdx.x) >> 5);
    int lane = threadIdx.x & 31;
    if (wave >= ntok) return;
    const float* xr = x + (size_t)wave * DIM;
    float acc[EXPERTS];
#pragma unroll
    for (int e = 0; e < EXPERTS; e++) acc[e] = 0.f;
    for (int d = lane; d < DIM; d += 32) {
        float xv = xr[d];
#pragma unroll
        for (int e = 0; e < EXPERTS; e++) acc[e] += xv * wr[e * DIM + d];
    }
#pragma unroll
    for (int m = 16; m >= 1; m >>= 1) {
#pragma unroll
        for (int e = 0; e < EXPERTS; e++) acc[e] += __shfl_xor(acc[e], m, 32);
    }
    if (lane == 0) {
        int bi = 0; float bv = acc[0];
#pragma unroll
        for (int e = 1; e < EXPERTS; e++)
            if (acc[e] > bv) { bv = acc[e]; bi = e; }   // strict >: first max, like jnp.argmax
        eid[wave] = bi;
        atomicAdd(&counts[bi], 1);
    }
}

// ---------------- kernel 4: scatter tokens into per-expert segments --------
__global__ void moe_scatter_kernel(const int* __restrict__ eid,
                                   int* __restrict__ cursors,
                                   int* __restrict__ list, int ntok) {
    int t = blockIdx.x * blockDim.x + threadIdx.x;
    if (t >= ntok) return;
    int e = eid[t];
    int p = atomicAdd(&cursors[e], 1);
    list[e * ntok + p] = t;
}

// ---------------- kernel 5: fused grouped GEMM (fc -> act -> proj) ---------
// 32 tokens per block; X streamed in two K=512 panels through Sbuf, then Sbuf
// is reused as the 32x512 bf16 activation buffer for the proj GEMM.
__global__ void __launch_bounds__(256, 1)
moe_gemm_kernel(const float* __restrict__ x,
                const unsigned short* __restrict__ wfc,   // [E][HID][DIM] bf16
                const unsigned short* __restrict__ wpj,   // [E][DIM][HID] bf16
                const int* __restrict__ counts,
                const int* __restrict__ list,
                float* __restrict__ out, int ntok, int tilesPerExpert) {
    __shared__ __align__(16) unsigned short Sbuf[MT * LSTR];  // ~33 KB, X-panel then H
    __shared__ int tokS[MT];

    int e    = blockIdx.x / tilesPerExpert;
    int tile = blockIdx.x % tilesPerExpert;
    int cnt  = counts[e];
    int base = tile * MT;
    if (base >= cnt) return;          // uniform exit, before any barrier/WMMA

    int tid = threadIdx.x;
    if (tid < MT) {
        int idx = base + tid;
        if (idx > cnt - 1) idx = cnt - 1;          // clamp: dup row, store masked later
        tokS[tid] = list[e * ntok + idx];
    }

    int wave  = tid >> 5;
    int lane  = tid & 31;
    int lhalf = lane >> 4;     // 0: lanes 0-15, 1: lanes 16-31
    int lmod  = lane & 15;

    // ================= FC: 32 token rows, 64 H cols per wave ===============
    const v8f vzero = {0.f, 0.f, 0.f, 0.f, 0.f, 0.f, 0.f, 0.f};
    v8f accF[2][4];
#pragma unroll
    for (int mt = 0; mt < 2; mt++)
#pragma unroll
        for (int j = 0; j < 4; j++) accF[mt][j] = vzero;
    int hbase = wave * 64;

    int frow = tid >> 3;               // 8 threads per row for panel fill
    int fcol = (tid & 7) * 64;

#pragma unroll
    for (int panel = 0; panel < 2; panel++) {
        __syncthreads();               // previous panel fully consumed (and tokS ready)
        // stage+convert X[:, panel*512 .. +512) -> Sbuf (bf16, padded rows)
        {
            const float4* src = (const float4*)(x + (size_t)tokS[frow] * DIM
                                                + panel * 512 + fcol);
            uint2* dst = (uint2*)(Sbuf + frow * LSTR + fcol);
#pragma unroll
            for (int i = 0; i < 16; i++) {
                float4 f = src[i];
                unsigned int lo = (unsigned int)f2bf(f.x) | ((unsigned int)f2bf(f.y) << 16);
                unsigned int hi = (unsigned int)f2bf(f.z) | ((unsigned int)f2bf(f.w) << 16);
                dst[i] = make_uint2(lo, hi);
            }
        }
        __syncthreads();

        for (int k = 0; k < 512; k += 32) {
            BFrag a0, a1;              // A fragments for the two 16-row M-tiles
            a0.q[0] = *(const uint4*)(Sbuf + lmod * LSTR + k + lhalf * 8);
            a0.q[1] = *(const uint4*)(Sbuf + lmod * LSTR + k + 16 + lhalf * 8);
            a1.q[0] = *(const uint4*)(Sbuf + (16 + lmod) * LSTR + k + lhalf * 8);
            a1.q[1] = *(const uint4*)(Sbuf + (16 + lmod) * LSTR + k + 16 + lhalf * 8);
#pragma unroll
            for (int j = 0; j < 4; j++) {
                int hrow = hbase + j * 16 + lmod;      // B col n = weight row h
                const uint4* bp = (const uint4*)(wfc + ((size_t)e * HID + hrow) * DIM
                                                     + panel * 512 + k + lhalf * 8);
                BFrag b; b.q[0] = bp[0]; b.q[1] = bp[2];
                accF[0][j] = __builtin_amdgcn_wmma_f32_16x16x32_bf16(
                    false, a0.bf, false, b.bf, (short)0, accF[0][j], false, false);
                accF[1][j] = __builtin_amdgcn_wmma_f32_16x16x32_bf16(
                    false, a1.bf, false, b.bf, (short)0, accF[1][j], false, false);
            }
        }
    }
    __syncthreads();                   // FC reads done; Sbuf becomes H buffer

    // activation: leaky_relu(0.5) then square; write bf16 H into Sbuf
#pragma unroll
    for (int mt = 0; mt < 2; mt++) {
#pragma unroll
        for (int j = 0; j < 4; j++) {
#pragma unroll
            for (int r = 0; r < 8; r++) {
                float v = accF[mt][j][r];
                v = (v >= 0.f) ? v : 0.5f * v;
                v = v * v;
                int m = mt * 16 + r + 8 * lhalf;       // C layout: m = vgpr + 8*(lane>=16)
                Sbuf[m * LSTR + hbase + j * 16 + lmod] = f2bf(v);
            }
        }
    }
    __syncthreads();

    // ================= PROJ: 128 D cols per wave, K = 512 ==================
    v8f accP[2][8];
#pragma unroll
    for (int mt = 0; mt < 2; mt++)
#pragma unroll
        for (int j = 0; j < 8; j++) accP[mt][j] = vzero;
    int dbase = wave * 128;

    for (int k = 0; k < HID; k += 32) {
        BFrag a0, a1;
        a0.q[0] = *(const uint4*)(Sbuf + lmod * LSTR + k + lhalf * 8);
        a0.q[1] = *(const uint4*)(Sbuf + lmod * LSTR + k + 16 + lhalf * 8);
        a1.q[0] = *(const uint4*)(Sbuf + (16 + lmod) * LSTR + k + lhalf * 8);
        a1.q[1] = *(const uint4*)(Sbuf + (16 + lmod) * LSTR + k + 16 + lhalf * 8);
#pragma unroll
        for (int j = 0; j < 8; j++) {
            int drow = dbase + j * 16 + lmod;          // B col n = weight row d
            const uint4* bp = (const uint4*)(wpj + ((size_t)e * DIM + drow) * HID
                                                 + k + lhalf * 8);
            BFrag b; b.q[0] = bp[0]; b.q[1] = bp[2];
            accP[0][j] = __builtin_amdgcn_wmma_f32_16x16x32_bf16(
                false, a0.bf, false, b.bf, (short)0, accP[0][j], false, false);
            accP[1][j] = __builtin_amdgcn_wmma_f32_16x16x32_bf16(
                false, a1.bf, false, b.bf, (short)0, accP[1][j], false, false);
        }
    }

    int valid = cnt - base; if (valid > MT) valid = MT;
#pragma unroll
    for (int mt = 0; mt < 2; mt++) {
#pragma unroll
        for (int j = 0; j < 8; j++) {
#pragma unroll
            for (int r = 0; r < 8; r++) {
                int m = mt * 16 + r + 8 * lhalf;
                if (m < valid) {
                    out[(size_t)tokS[m] * DIM + dbase + j * 16 + lmod] = accP[mt][j][r];
                }
            }
        }
    }
}

// ===========================================================================
extern "C" void kernel_launch(void* const* d_in, const int* in_sizes, int n_in,
                              void* d_out, int out_size, void* d_ws, size_t ws_size,
                              hipStream_t stream) {
    const float* x   = (const float*)d_in[0];
    const float* wr  = (const float*)d_in[1];
    const float* wfc = (const float*)d_in[2];
    const float* wpj = (const float*)d_in[3];
    float* out = (float*)d_out;

    const int ntok = in_sizes[0] / DIM;          // 4*4096 = 16384

    // workspace layout
    char* ws = (char*)d_ws;
    int* eid     = (int*)ws;                                   // ntok ints
    size_t off   = (size_t)ntok * 4;
    int* counts  = (int*)(ws + off);                           // 8 ints
    int* cursors = counts + 8;                                 // 8 ints
    off += 64;
    int* list    = (int*)(ws + off);                           // EXPERTS*ntok ints
    off += (size_t)EXPERTS * ntok * 4;
    off  = (off + 255) & ~(size_t)255;
    unsigned short* wfcB = (unsigned short*)(ws + off);        // 8 MB bf16
    off += (size_t)EXPERTS * HID * DIM * 2;
    unsigned short* wpjB = (unsigned short*)(ws + off);        // 8 MB bf16

    // 1) zero counters (ws is not re-initialized between replays)
    moe_init_kernel<<<1, 32, 0, stream>>>(counts);

    // 2) weight fp32 -> bf16 (one cheap pass)
    {
        int n4 = EXPERTS * HID * DIM / 4;
        moe_cvt_kernel<<<(n4 + 255) / 256, 256, 0, stream>>>(wfc, wfcB, n4);
        moe_cvt_kernel<<<(n4 + 255) / 256, 256, 0, stream>>>(wpj, wpjB, n4);
    }

    // 3) router: one wave per token
    moe_router_kernel<<<(ntok * 32 + 255) / 256, 256, 0, stream>>>(x, wr, eid, counts, ntok);

    // 4) scatter tokens into per-expert segments
    moe_scatter_kernel<<<(ntok + 255) / 256, 256, 0, stream>>>(eid, cursors, list, ntok);

    // 5) fused grouped GEMM; worst-case static grid (graph-capture friendly)
    {
        int tilesPerExpert = ntok / MT;          // 512
        dim3 grid(EXPERTS * tilesPerExpert);     // 4096 blocks, empties exit early
        moe_gemm_kernel<<<grid, 256, 0, stream>>>(x, wfcB, wpjB, counts, list,
                                                  out, ntok, tilesPerExpert);
    }
}
